// SpatialMSA_76106820485278
// MI455X (gfx1250) — matrix operand order repaired and checked by reading, not compiled, not run on hardware
//
#include <hip/hip_runtime.h>

// ---------------------------------------------------------------------------
// SpatialMSA on MI455X (gfx1250): bf16 WMMA everywhere, f32 accumulate.
// Pipeline: [1] patch-gather + qkv GEMM  [2] per-(n,head) attention
//           [3] output projection GEMM + scatter.
// Contiguous LDS staging uses CDNA5 async global->LDS DMA (ASYNCcnt).
// ---------------------------------------------------------------------------

typedef __attribute__((ext_vector_type(16))) __bf16 v16bf;
typedef __attribute__((ext_vector_type(8)))  float  v8f;
typedef unsigned short ushort_t;
typedef unsigned int   uint_t;

union Frag16 { v16bf v; uint_t u[8]; };

__device__ __forceinline__ ushort_t f2bf(float f) {
  uint_t u = __float_as_uint(f);
  u += 0x7FFFu + ((u >> 16) & 1u);          // round-to-nearest-even
  return (ushort_t)(u >> 16);
}
__device__ __forceinline__ float bf2f(ushort_t h) {
  return __uint_as_float(((uint_t)h) << 16);
}
__device__ __forceinline__ v8f zero8() {
  v8f z = {0.f, 0.f, 0.f, 0.f, 0.f, 0.f, 0.f, 0.f};
  return z;
}
__device__ __forceinline__ v8f wmma_bf16(v16bf a, v16bf b, v8f c) {
  return __builtin_amdgcn_wmma_f32_16x16x32_bf16(false, a, false, b, (short)0, c,
                                                 false, false);
}

// CDNA5 async copy: 16 bytes/lane, global -> LDS, tracked by ASYNCcnt.
// lds_off is the wave-relative LDS byte address (low 32 bits of an LDS
// generic pointer); hardware adds LDS_BASE.
__device__ __forceinline__ void async_copy_b128(uint_t lds_off, const void* g) {
  asm volatile("global_load_async_to_lds_b128 %0, %1, off"
               :: "v"(lds_off), "v"(g) : "memory");
}
__device__ __forceinline__ void wait_async0() {
  asm volatile("s_wait_asynccnt 0x0" ::: "memory");
}

// A fragment (16x32 bf16) from a [M][K] buffer, K contiguous, 2 bf16/dword.
// CDNA5 wave32 A-layout: lane L -> row (caller passes m = base + (L&15));
// vgpr v -> K = (v>=4?16:0) + (L>>4)*8 + (v&3)*2.
__device__ __forceinline__ v16bf load_fragA(const uint_t* p32, int m,
                                            int stride32, int kbase, int lane) {
  const int g = lane >> 4;
  Frag16 f;
#pragma unroll
  for (int v = 0; v < 8; ++v) {
    int kk = ((v >= 4) ? 16 : 0) + g * 8 + (v & 3) * 2;
    f.u[v] = p32[m * stride32 + ((kbase + kk) >> 1)];
  }
  return f.v;
}
// B fragment (32x16 bf16) from a [N][K] buffer (K contiguous).
// B-layout: lane L -> col (caller passes ncol); vgpr v -> K = (v>=4?16:0) + (L>>4)*8 + (v&3)*2.
__device__ __forceinline__ v16bf load_fragB(const uint_t* p32, int ncol,
                                            int stride32, int kbase, int lane) {
  const int g = lane >> 4;
  Frag16 f;
#pragma unroll
  for (int v = 0; v < 8; ++v) {
    int kk = ((v >= 4) ? 16 : 0) + g * 8 + (v & 3) * 2;
    f.u[v] = p32[ncol * stride32 + ((kbase + kk) >> 1)];
  }
  return f.v;
}

// ---------------------------------------------------------------------------
// Kernel 1: qkv = t @ w_in^T + b_in  (t gathered from x patches), bf16 out.
// Grid: (nIdx, cchunk) = 512*4 blocks, 256 threads (8 waves).
// Output layout: qkv[((n*3+which)*4+h)*256 + c]*64 + d, bf16; q pre-scaled.
// (Staging keeps the VALU path: fp32 -> bf16 conversion happens in flight.)
// ---------------------------------------------------------------------------
__global__ __launch_bounds__(256) void qkv_proj_kernel(
    const float* __restrict__ x, const float* __restrict__ w_in,
    const float* __restrict__ b_in, ushort_t* __restrict__ qkv) {
  extern __shared__ char smem[];
  ushort_t* Asm = (ushort_t*)smem;            // [64 c][256 e]  bf16
  ushort_t* Bsm = (ushort_t*)(smem + 32768);  // [64 ncol][256 k] bf16
  const int tid = threadIdx.x;
  const int lane = tid & 31, wave = tid >> 5;
  const int cc = blockIdx.x & 3;
  const int nIdx = blockIdx.x >> 2;
  const int b = nIdx >> 8, p = nIdx & 255;
  const int ph = p >> 4, pw = p & 15;
  const int c0 = cc * 64;

  // Stage A once: t[n, c0..c0+63, 0..255] from x (coalesced float4 reads).
#pragma unroll
  for (int it = 0; it < 16; ++it) {
    int f = (it * 256 + tid) * 4;           // element index in 64x256 tile
    int r = f >> 8, e = f & 255;
    int i = e >> 4, j = e & 15;
    float4 val = *(const float4*)(
        x + ((((size_t)(b * 256 + c0 + r)) * 256 + ph * 16 + i) * 256 +
             pw * 16 + j));
    uint_t* dst = (uint_t*)(Asm + r * 256 + e);
    dst[0] = (uint_t)f2bf(val.x) | ((uint_t)f2bf(val.y) << 16);
    dst[1] = (uint_t)f2bf(val.z) | ((uint_t)f2bf(val.w) << 16);
  }

  const uint_t* A32 = (const uint_t*)Asm;
  const uint_t* B32 = (const uint_t*)Bsm;
  const int rt = wave >> 1;          // 16-row tile of the 64-row chunk
  const int ct0 = (wave & 1) * 2;    // first of two 16-col tiles
  const int m = rt * 16 + (lane & 15);
  const int nc = tid >> 2, kq = tid & 3;

  for (int colc = 0; colc < 12; ++colc) {   // 12 * 64 = 768 output cols
    __syncthreads();
    // Stage B chunk: B[ncol][k] = w_in[colc*64+ncol][k] as bf16.
    const float* wrow = w_in + (size_t)(colc * 64 + nc) * 256 + kq * 64;
    uint_t* brow = (uint_t*)(Bsm + nc * 256 + kq * 64);
#pragma unroll
    for (int q = 0; q < 16; ++q) {
      float4 w4 = *(const float4*)(wrow + q * 4);
      brow[q * 2] = (uint_t)f2bf(w4.x) | ((uint_t)f2bf(w4.y) << 16);
      brow[q * 2 + 1] = (uint_t)f2bf(w4.z) | ((uint_t)f2bf(w4.w) << 16);
    }
    __syncthreads();

    v8f acc0 = zero8(), acc1 = zero8();
#pragma unroll
    for (int ks = 0; ks < 256; ks += 32) {
      v16bf a = load_fragA(A32, m, 128, ks, lane);
      acc0 = wmma_bf16(a, load_fragB(B32, ct0 * 16 + (lane & 15), 128, ks, lane), acc0);
      acc1 = wmma_bf16(a, load_fragB(B32, ct0 * 16 + 16 + (lane & 15), 128, ks, lane), acc1);
    }

#pragma unroll
    for (int cti = 0; cti < 2; ++cti) {
      v8f acc = cti ? acc1 : acc0;
      int gcol = colc * 64 + (ct0 + cti) * 16 + (lane & 15);
      float bias = b_in[gcol];
      int which = gcol >> 8;                 // 0=q,1=k,2=v
      int e = gcol & 255;
      int h = e >> 6, d = e & 63;
      float scale = (which == 0) ? 0.125f : 1.0f;   // fold HD^-0.5 into q
      int rowb = rt * 16 + ((lane >> 4) * 8);
#pragma unroll
      for (int r = 0; r < 8; ++r) {
        int c = c0 + rowb + r;
        float v = (acc[r] + bias) * scale;
        qkv[((((size_t)nIdx * 3 + which) * 4 + h) * 256 + c) * 64 + d] = f2bf(v);
      }
    }
  }
}

// ---------------------------------------------------------------------------
// Kernel 2: per-(n, head) attention. Grid: 512*4 blocks, 256 threads.
// LDS: K[256][64] bf16 (32K) + V^T[64][256] bf16 (32K) + per-wave exp tile
// (8 * 32x32 bf16 = 16K) = 80 KB.
// K tile staged with async global->LDS DMA; V transposed via VALU path.
// Each wave owns 32 query rows; q fragments in registers; 2-pass softmax.
// ---------------------------------------------------------------------------
__global__ __launch_bounds__(256) void attn_kernel(
    const ushort_t* __restrict__ qkv, ushort_t* __restrict__ ows) {
  extern __shared__ char smem[];
  ushort_t* Kt = (ushort_t*)smem;             // [key][d]  -> B-layout (N=key,K=d)
  ushort_t* Vt = (ushort_t*)(smem + 32768);   // [d][key]  -> B-layout (N=d,K=key)
  ushort_t* SE = (ushort_t*)(smem + 65536);   // per-wave [32][32] exp tiles
  const int tid = threadIdx.x, lane = tid & 31, wave = tid >> 5;
  const int nIdx = blockIdx.x >> 2, h = blockIdx.x & 3;
  const ushort_t* qg = qkv + (((size_t)nIdx * 3 + 0) * 4 + h) * 16384;
  const ushort_t* kg = qkv + (((size_t)nIdx * 3 + 1) * 4 + h) * 16384;
  const ushort_t* vg = qkv + (((size_t)nIdx * 3 + 2) * 4 + h) * 16384;

  {  // Stage K tile: async DMA straight into LDS (32 KB, no VGPR bounce).
    const uint_t kt_off = (uint_t)(uintptr_t)Kt;
#pragma unroll
    for (int i = 0; i < 8; ++i) {
      int idx = i * 256 + tid;
      async_copy_b128(kt_off + (uint_t)idx * 16, (const void*)(kg + (size_t)idx * 8));
    }
  }
  {  // Stage V transposed: thread = key row, scatter into [d][key].
    const uint4* src = (const uint4*)(vg + (size_t)tid * 64);
#pragma unroll
    for (int q = 0; q < 8; ++q) {
      uint4 v4 = src[q];
      const ushort_t* pv = (const ushort_t*)&v4;
#pragma unroll
      for (int t = 0; t < 8; ++t) Vt[(q * 8 + t) * 256 + tid] = pv[t];
    }
  }

  // q fragments straight from global (rows = this wave's 32 queries),
  // overlapped with the async K copy still in flight.
  const uint_t* q32 = (const uint_t*)qg;
  const int c0 = wave * 32;
  v16bf qa[2][2];
#pragma unroll
  for (int rt = 0; rt < 2; ++rt)
#pragma unroll
    for (int ks = 0; ks < 2; ++ks)
      qa[rt][ks] = load_fragA(q32, c0 + rt * 16 + (lane & 15), 32, ks * 32, lane);

  wait_async0();        // this wave's async K writes complete
  __syncthreads();      // everyone's writes visible

  const uint_t* K32 = (const uint_t*)Kt;
  const uint_t* V32 = (const uint_t*)Vt;
  ushort_t* SEw = SE + wave * 1024;
  const uint_t* SE32 = (const uint_t*)SEw;

  float runm[2][8];
#pragma unroll
  for (int rt = 0; rt < 2; ++rt)
#pragma unroll
    for (int r = 0; r < 8; ++r) runm[rt][r] = -3.0e38f;

  // ---- Pass 1: scores -> running row maxima (register only) ----
  for (int jt = 0; jt < 8; ++jt) {
#pragma unroll
    for (int rt = 0; rt < 2; ++rt) {
#pragma unroll
      for (int ns = 0; ns < 2; ++ns) {
        v8f acc = zero8();
        int key = jt * 32 + ns * 16 + (lane & 15);
#pragma unroll
        for (int ks = 0; ks < 2; ++ks)
          acc = wmma_bf16(qa[rt][ks], load_fragB(K32, key, 32, ks * 32, lane), acc);
#pragma unroll
        for (int r = 0; r < 8; ++r) runm[rt][r] = fmaxf(runm[rt][r], acc[r]);
      }
    }
  }
  // reduce maxima across the 16 lanes holding one row
#pragma unroll
  for (int rt = 0; rt < 2; ++rt)
#pragma unroll
    for (int r = 0; r < 8; ++r) {
      float mv = runm[rt][r];
      mv = fmaxf(mv, __shfl_xor(mv, 1, 32));
      mv = fmaxf(mv, __shfl_xor(mv, 2, 32));
      mv = fmaxf(mv, __shfl_xor(mv, 4, 32));
      mv = fmaxf(mv, __shfl_xor(mv, 8, 32));
      runm[rt][r] = mv;
    }

  // ---- Pass 2: recompute scores, exp, P@V, row sums ----
  v8f oacc[2][4];
#pragma unroll
  for (int rt = 0; rt < 2; ++rt)
#pragma unroll
    for (int dt = 0; dt < 4; ++dt) oacc[rt][dt] = zero8();
  float psum[2][8];
#pragma unroll
  for (int rt = 0; rt < 2; ++rt)
#pragma unroll
    for (int r = 0; r < 8; ++r) psum[rt][r] = 0.f;

  for (int jt = 0; jt < 8; ++jt) {
#pragma unroll
    for (int rt = 0; rt < 2; ++rt) {
#pragma unroll
      for (int ns = 0; ns < 2; ++ns) {
        v8f acc = zero8();
        int key = jt * 32 + ns * 16 + (lane & 15);
#pragma unroll
        for (int ks = 0; ks < 2; ++ks)
          acc = wmma_bf16(qa[rt][ks], load_fragB(K32, key, 32, ks * 32, lane), acc);
        int rowb = (lane >> 4) * 8;
#pragma unroll
        for (int r = 0; r < 8; ++r) {
          float ev = __expf(acc[r] - runm[rt][r]);
          psum[rt][r] += ev;
          // C-layout -> LDS staging tile (per wave; DS ops in-order per wave)
          SEw[(rt * 16 + rowb + r) * 32 + ns * 16 + (lane & 15)] = f2bf(ev);
        }
      }
    }
    // reload exp tile as A fragments, accumulate o += P_tile @ V_tile
    v16bf af0 = load_fragA(SE32, (lane & 15), 16, 0, lane);
    v16bf af1 = load_fragA(SE32, 16 + (lane & 15), 16, 0, lane);
#pragma unroll
    for (int dt = 0; dt < 4; ++dt) {
      v16bf bv = load_fragB(V32, dt * 16 + (lane & 15), 128, jt * 32, lane);
      oacc[0][dt] = wmma_bf16(af0, bv, oacc[0][dt]);
      oacc[1][dt] = wmma_bf16(af1, bv, oacc[1][dt]);
    }
  }

  // finalize row sums, normalize, store o (bf16) to workspace [n][c][E]
#pragma unroll
  for (int rt = 0; rt < 2; ++rt)
#pragma unroll
    for (int r = 0; r < 8; ++r) {
      float s = psum[rt][r];
      s += __shfl_xor(s, 1, 32);
      s += __shfl_xor(s, 2, 32);
      s += __shfl_xor(s, 4, 32);
      s += __shfl_xor(s, 8, 32);
      psum[rt][r] = 1.0f / s;
    }
#pragma unroll
  for (int rt = 0; rt < 2; ++rt) {
    int rowb = rt * 16 + (lane >> 4) * 8;
#pragma unroll
    for (int r = 0; r < 8; ++r) {
      int cq = c0 + rowb + r;
      size_t obase = ((size_t)nIdx * 256 + cq) * 256 + h * 64;
      float inv = psum[rt][r];
#pragma unroll
      for (int dt = 0; dt < 4; ++dt)
        ows[obase + dt * 16 + (lane & 15)] = f2bf(oacc[rt][dt][r] * inv);
    }
  }
}

// ---------------------------------------------------------------------------
// Kernel 3: out = o @ w_out^T + b_out, scattered to (B, C, nP, E) f32.
// Grid: 131072/64 = 2048 blocks, 256 threads. A tile staged via async DMA.
// ---------------------------------------------------------------------------
__global__ __launch_bounds__(256) void out_proj_kernel(
    const ushort_t* __restrict__ ows, const float* __restrict__ w_out,
    const float* __restrict__ b_out, float* __restrict__ out) {
  extern __shared__ char smem[];
  ushort_t* Asm = (ushort_t*)smem;            // [64 rows][256 k]
  ushort_t* Bsm = (ushort_t*)(smem + 32768);  // [64 ncol][256 k]
  const int tid = threadIdx.x, lane = tid & 31, wave = tid >> 5;
  const size_t row0 = (size_t)blockIdx.x * 64;

  {  // stage A: 64 contiguous bf16 rows of o via async global->LDS DMA
    const uint_t a_off = (uint_t)(uintptr_t)Asm;
    const ushort_t* src = ows + row0 * 256;
#pragma unroll
    for (int i = 0; i < 8; ++i) {
      int idx = i * 256 + tid;
      async_copy_b128(a_off + (uint_t)idx * 16, (const void*)(src + (size_t)idx * 8));
    }
    wait_async0();
  }
  const uint_t* A32 = (const uint_t*)Asm;
  const uint_t* B32 = (const uint_t*)Bsm;
  const int rt = wave >> 1, ct0 = (wave & 1) * 2;
  const int m = rt * 16 + (lane & 15);
  const int nc = tid >> 2, kq = tid & 3;

  for (int colc = 0; colc < 4; ++colc) {
    __syncthreads();
    const float* wrow = w_out + (size_t)(colc * 64 + nc) * 256 + kq * 64;
    uint_t* brow = (uint_t*)(Bsm + nc * 256 + kq * 64);
#pragma unroll
    for (int q = 0; q < 16; ++q) {
      float4 w4 = *(const float4*)(wrow + q * 4);
      brow[q * 2] = (uint_t)f2bf(w4.x) | ((uint_t)f2bf(w4.y) << 16);
      brow[q * 2 + 1] = (uint_t)f2bf(w4.z) | ((uint_t)f2bf(w4.w) << 16);
    }
    __syncthreads();

    v8f acc0 = zero8(), acc1 = zero8();
#pragma unroll
    for (int ks = 0; ks < 256; ks += 32) {
      v16bf a = load_fragA(A32, m, 128, ks, lane);
      acc0 = wmma_bf16(a, load_fragB(B32, ct0 * 16 + (lane & 15), 128, ks, lane), acc0);
      acc1 = wmma_bf16(a, load_fragB(B32, ct0 * 16 + 16 + (lane & 15), 128, ks, lane), acc1);
    }

#pragma unroll
    for (int cti = 0; cti < 2; ++cti) {
      v8f acc = cti ? acc1 : acc0;
      int gcol = colc * 64 + (ct0 + cti) * 16 + (lane & 15);
      float bias = b_out[gcol];
      int rowb = rt * 16 + (lane >> 4) * 8;
#pragma unroll
      for (int r = 0; r < 8; ++r) {
        size_t grow = row0 + rowb + r;      // = n*256 + c
        size_t nIdx = grow >> 8;
        int c = (int)(grow & 255);
        size_t bb = nIdx >> 8;
        size_t p = nIdx & 255;
        out[(((bb * 256 + c) * 256 + p) * 256) + gcol] = acc[r] + bias;
      }
    }
  }
}

// ---------------------------------------------------------------------------
extern "C" void kernel_launch(void* const* d_in, const int* in_sizes, int n_in,
                              void* d_out, int out_size, void* d_ws,
                              size_t ws_size, hipStream_t stream) {
  (void)in_sizes; (void)n_in; (void)out_size; (void)ws_size;
  const float* x = (const float*)d_in[0];
  const float* w_in = (const float*)d_in[1];
  const float* b_in = (const float*)d_in[2];
  const float* w_out = (const float*)d_in[3];
  const float* b_out = (const float*)d_in[4];
  float* out = (float*)d_out;

  // workspace: qkv (512*3*4*256*64 bf16 = 201 MB) then o (512*256*256 bf16 = 67 MB)
  ushort_t* qkv = (ushort_t*)d_ws;
  ushort_t* ows = qkv + (size_t)512 * 196608;

  qkv_proj_kernel<<<dim3(2048), dim3(256), 65536, stream>>>(x, w_in, b_in, qkv);
  attn_kernel<<<dim3(2048), dim3(256), 81920, stream>>>(qkv, ows);
  out_proj_kernel<<<dim3(2048), dim3(256), 65536, stream>>>(ows, w_out, b_out, out);
}
